// RoSTER_48515950576035
// MI455X (gfx1250) — matrix operand order repaired and checked by reading, not compiled, not run on hardware
//
#include <hip/hip_runtime.h>
#include <math.h>

// Problem constants (from reference): B=64, S=512, E=768, H=768, C=9
constexpr int   Ecst   = 768;
constexpr int   Hcst   = 768;
constexpr int   Ccst   = 9;
constexpr int   M_TOT  = 64 * 512;     // 32768 tokens
constexpr int   M_TILE = 128;          // rows per workgroup
constexpr int   N_TILE = 64;           // W1 columns per chunk
constexpr int   K_TILE = 32;           // WMMA K
constexpr int   KITERS = Ecst / K_TILE; // 24
constexpr float TAU    = 0.7f;
constexpr float QGCE   = 0.7f;
constexpr int   IGNORE = -100;

typedef __attribute__((ext_vector_type(16))) __bf16          v16bf;
typedef __attribute__((ext_vector_type(8)))  float           v8f;
typedef __attribute__((ext_vector_type(8)))  unsigned short  us8;   // 16B
typedef __attribute__((ext_vector_type(4)))  __bf16          bf4;   // 8B

union ABFrag { v16bf v; us8 h[2]; };

__global__ __launch_bounds__(256)
void roster_head_fused(const float* __restrict__ hidden,
                       const float* __restrict__ W1,
                       const float* __restrict__ b1,
                       const float* __restrict__ W2,
                       const float* __restrict__ b2,
                       const int*   __restrict__ labels,
                       const int*   __restrict__ epoch_p,
                       float* __restrict__ probs,
                       float* __restrict__ loss)
{
  __shared__ __attribute__((aligned(16))) unsigned short a_lds[M_TILE * K_TILE]; //  8 KB  A: [row][k]
  __shared__ __attribute__((aligned(16))) unsigned short b_lds[N_TILE * K_TILE]; //  4 KB  W1ᵀ: [n][k]
  __shared__ __attribute__((aligned(16))) unsigned short h_lds[M_TILE * N_TILE]; // 16 KB  h chunk: [row][k]
  __shared__ __attribute__((aligned(16))) unsigned short w2_lds[16 * Hcst];      // 24 KB  W2ᵀ padded: [n(16)][k]

  const int t    = threadIdx.x;
  const int lane = t & 31;
  const int wv   = t >> 5;            // wave 0..7
  const int m0   = blockIdx.x * M_TILE;
  const int r0   = wv * 16;           // wave's 16 rows inside the M tile
  const int lrow = lane & 15;
  const int hi   = (lane >> 4) & 1;   // lane half

  // Per-thread staging coordinates (fixed for the whole kernel)
  int a_row[4], a_kq[4], b_k[2], b_nq[2];
  #pragma unroll
  for (int i = 0; i < 4; ++i) { int f = i * 256 + t; a_row[i] = f >> 3; a_kq[i] = (f & 7) << 2; }
  #pragma unroll
  for (int i = 0; i < 2; ++i) { int f = i * 256 + t; b_k[i] = f >> 4;  b_nq[i] = (f & 15) << 2; }

  // ---- Stage W2 transposed & zero-padded (9 -> 16 cols) once ----
  for (int idx = t; idx < 16 * Hcst; idx += 256) {
    int n = idx / Hcst, k = idx - n * Hcst;
    float v = (n < Ccst) ? W2[(size_t)k * Ccst + n] : 0.f;
    *(__bf16*)(w2_lds + n * Hcst + k) = (__bf16)v;     // hw f32->bf16 convert
  }

  v8f acc2 = {0.f, 0.f, 0.f, 0.f, 0.f, 0.f, 0.f, 0.f};   // persistent 16x16 logits

  float4 sa[4], sb[2];                 // register-staged next tile

  for (int nt = 0; nt < Hcst / N_TILE; ++nt) {
    const int n0 = nt * N_TILE;
    v8f z = {0.f, 0.f, 0.f, 0.f, 0.f, 0.f, 0.f, 0.f};
    v8f acc1[4] = {z, z, z, z};

    // Prologue: issue global loads for k-chunk 0 (no wait yet)
    #pragma unroll
    for (int i = 0; i < 4; ++i)
      sa[i] = *(const float4*)(hidden + (size_t)(m0 + a_row[i]) * Ecst + a_kq[i]);
    #pragma unroll
    for (int i = 0; i < 2; ++i)
      sb[i] = *(const float4*)(W1 + (size_t)b_k[i] * Hcst + n0 + b_nq[i]);

    for (int kk = 0; kk < KITERS; ++kk) {
      __syncthreads();   // prior iteration's fragment reads done; LDS reusable

      // ---- Store staged registers -> LDS (hardware bf16 converts) ----
      #pragma unroll
      for (int i = 0; i < 4; ++i) {
        bf4 p = { (__bf16)sa[i].x, (__bf16)sa[i].y, (__bf16)sa[i].z, (__bf16)sa[i].w };
        *(bf4*)(a_lds + a_row[i] * K_TILE + a_kq[i]) = p;
      }
      #pragma unroll
      for (int i = 0; i < 2; ++i) {
        *(__bf16*)(b_lds + (b_nq[i] + 0) * K_TILE + b_k[i]) = (__bf16)sb[i].x;
        *(__bf16*)(b_lds + (b_nq[i] + 1) * K_TILE + b_k[i]) = (__bf16)sb[i].y;
        *(__bf16*)(b_lds + (b_nq[i] + 2) * K_TILE + b_k[i]) = (__bf16)sb[i].z;
        *(__bf16*)(b_lds + (b_nq[i] + 3) * K_TILE + b_k[i]) = (__bf16)sb[i].w;
      }

      // ---- Issue next k-chunk's global loads; latency hides under WMMAs ----
      if (kk + 1 < KITERS) {
        const int k1 = (kk + 1) * K_TILE;
        #pragma unroll
        for (int i = 0; i < 4; ++i)
          sa[i] = *(const float4*)(hidden + (size_t)(m0 + a_row[i]) * Ecst + k1 + a_kq[i]);
        #pragma unroll
        for (int i = 0; i < 2; ++i)
          sb[i] = *(const float4*)(W1 + (size_t)(k1 + b_k[i]) * Hcst + n0 + b_nq[i]);
      }

      __syncthreads();   // staged tile visible to all waves

      // ---- A fragment (16x32): two ds_load_b128 per lane ----
      ABFrag af;
      const int am    = r0 + lrow;
      const int abase = hi ? 8 : 0;
      af.h[0] = *(const us8*)(a_lds + am * K_TILE + abase);
      af.h[1] = *(const us8*)(a_lds + am * K_TILE + abase + 16);

      #pragma unroll
      for (int tt = 0; tt < 4; ++tt) {    // 4 N-subtiles of 16
        ABFrag bf;
        const int bn    = tt * 16 + lrow;
        const int bbase = hi ? 8 : 0;
        bf.h[0] = *(const us8*)(b_lds + bn * K_TILE + bbase);
        bf.h[1] = *(const us8*)(b_lds + bn * K_TILE + bbase + 16);
        acc1[tt] = __builtin_amdgcn_wmma_f32_16x16x32_bf16(
            false, af.v, false, bf.v, (short)0, acc1[tt], false, false);
      }
    }

    // ---- bias + tanh, store h chunk (wave-private rows -> no barrier needed) ----
    #pragma unroll
    for (int tt = 0; tt < 4; ++tt) {
      const int col  = tt * 16 + lrow;
      const float bv = b1[n0 + col];
      #pragma unroll
      for (int j = 0; j < 8; ++j) {
        const int row = r0 + j + (hi ? 8 : 0);
        *(__bf16*)(h_lds + row * N_TILE + col) = (__bf16)tanhf(acc1[tt][j] + bv);
      }
    }

    // ---- GEMM2: logits += h_chunk(16x64) x W2ᵀ-pad(64x16), two WMMAs ----
    #pragma unroll
    for (int k2 = 0; k2 < N_TILE; k2 += K_TILE) {
      ABFrag af2;
      const int am = r0 + lrow;
      const int kb = k2 + (hi ? 8 : 0);
      af2.h[0] = *(const us8*)(h_lds + am * N_TILE + kb);
      af2.h[1] = *(const us8*)(h_lds + am * N_TILE + kb + 16);
      ABFrag bf2;
      const int ks = n0 + k2 + (hi ? 8 : 0);
      bf2.h[0] = *(const us8*)(w2_lds + lrow * Hcst + ks);
      bf2.h[1] = *(const us8*)(w2_lds + lrow * Hcst + ks + 16);
      acc2 = __builtin_amdgcn_wmma_f32_16x16x32_bf16(
          false, af2.v, false, bf2.v, (short)0, acc2, false, false);
    }
  }

  // ---- softmax over C (16-lane groups), gather, GCE loss ----
  const int epoch = epoch_p[0];
  const int c     = lrow;
  float wave_loss = 0.f;

  #pragma unroll
  for (int j = 0; j < 8; ++j) {
    const int m = m0 + r0 + j + (hi ? 8 : 0);
    float lg = (c < Ccst) ? (acc2[j] + b2[c]) : -INFINITY;
    float mx = lg;
    #pragma unroll
    for (int off = 8; off >= 1; off >>= 1)
      mx = fmaxf(mx, __shfl_xor(mx, off, 16));
    float e = (c < Ccst) ? __expf(lg - mx) : 0.f;
    float s = e;
    #pragma unroll
    for (int off = 8; off >= 1; off >>= 1)
      s += __shfl_xor(s, off, 16);
    float p = e / s;
    if (c < Ccst) probs[(size_t)m * Ccst + c] = p;

    const int lab = labels[m];
    float pl = (c == lab) ? p : 0.f;
    #pragma unroll
    for (int off = 8; off >= 1; off >>= 1)
      pl += __shfl_xor(pl, off, 16);
    if (c == 0 && lab != IGNORE) {
      float w = (epoch <= 2) ? 1.f : ((pl > TAU) ? 1.f : 0.f);
      wave_loss += w * (1.f - __powf(pl, QGCE)) * (1.f / QGCE);
    }
  }

  #pragma unroll
  for (int off = 16; off >= 1; off >>= 1)
    wave_loss += __shfl_xor(wave_loss, off, 32);
  if (lane == 0 && wave_loss != 0.f)
    atomicAdd(loss, wave_loss);
}

extern "C" void kernel_launch(void* const* d_in, const int* in_sizes, int n_in,
                              void* d_out, int out_size, void* d_ws, size_t ws_size,
                              hipStream_t stream) {
  const float* hidden = (const float*)d_in[0];
  const float* W1     = (const float*)d_in[1];
  const float* b1     = (const float*)d_in[2];
  const float* W2     = (const float*)d_in[3];
  const float* b2     = (const float*)d_in[4];
  const int*   labels = (const int*)d_in[5];
  const int*   epoch  = (const int*)d_in[6];

  float* probs = (float*)d_out;
  float* loss  = probs + (size_t)M_TOT * Ccst;   // out = [probs | loss]

  hipMemsetAsync(loss, 0, sizeof(float), stream);

  dim3 grid(M_TOT / M_TILE);   // 256 workgroups
  roster_head_fused<<<grid, 256, 0, stream>>>(hidden, W1, b1, W2, b2,
                                              labels, epoch, probs, loss);
}
